// MambaBlock_84722524881192
// MI455X (gfx1250) — matrix operand order repaired and checked by reading, not compiled, not run on hardware
//
#include <hip/hip_runtime.h>
#include <hip/hip_bf16.h>

// ---------------------------------------------------------------------------
// Bi-directional Mamba block for MI455X (gfx1250, wave32, WMMA + TDM).
// Pipeline per direction (f, b):
//   1) xz = x @ in_w^T              -> WMMA f16 GEMM (split into xc | z)
//   2) u  = silu(depthwise_conv(xc))-> elementwise
//   3) xdb = u @ xproj_w^T          -> WMMA f16 GEMM (N=48)
//   4) selective scan: TDM streams xdb rows into LDS in 128-step chunks
//      (double buffered, s_wait_tensorcnt), recurrence in registers
//   5) ym = g @ out_w^T             -> WMMA f16 GEMM
//   6) out = rmsnorm(ym + x), time-flip for the backward direction
// ---------------------------------------------------------------------------

typedef __attribute__((ext_vector_type(8)))  _Float16 v8h;
typedef __attribute__((ext_vector_type(16))) _Float16 v16h;
typedef __attribute__((ext_vector_type(8)))  float    v8f;
typedef __attribute__((ext_vector_type(4)))  unsigned u32x4;
typedef __attribute__((ext_vector_type(4)))  int      i32x4;
typedef __attribute__((ext_vector_type(8)))  int      i32x8;

#define BATCH 2
#define SEQ   2048
#define DM    256
#define DI    1024
#define NS    16
#define BS    (BATCH * SEQ)   // 4096 rows per direction
#define CHUNK 128             // scan timesteps per TDM tile

// --------------------------- TDM availability ------------------------------
#if defined(__has_builtin)
#  if __has_builtin(__builtin_amdgcn_tensor_load_to_lds) && \
      __has_builtin(__builtin_amdgcn_s_wait_tensorcnt)
#    define MAMBA_TDM 1
#  endif
#endif
#ifndef MAMBA_TDM
#  define MAMBA_TDM 0
#  warning "MAMBA: TDM builtins unavailable; falling back to cooperative LDS copy"
#endif

// ---------------------------------------------------------------------------
// WMMA fragment helpers (wave32, 16x16x32 f16).
// LDS tiles are row-major [rows][32 halves], row stride 32 halves (64B).
// A frag: lane (hi,lo): M = lo, halves i   -> K = hi*16+i
// B frag: lane (hi,lo): N = lo, halves i   -> K = hi*16+i
// ---------------------------------------------------------------------------
static __device__ __forceinline__ v16h frag_cat(v8h a, v8h b) {
  v16h r;
#pragma unroll
  for (int i = 0; i < 8; ++i) { r[i] = a[i]; r[i + 8] = b[i]; }
  return r;
}

static __device__ __forceinline__ v16h frag_A(const _Float16* lds, int rowbase, int lane) {
  const int lo = lane & 15, hi = lane >> 4;
  const _Float16* row = lds + (rowbase + lo) * 32;
  v8h p0 = *(const v8h*)(row + hi * 16);
  v8h p1 = *(const v8h*)(row + hi * 16 + 8);
  return frag_cat(p0, p1);
}

static __device__ __forceinline__ v16h frag_B(const _Float16* lds, int colbase, int lane) {
  const int lo = lane & 15, hi = lane >> 4;
  const _Float16* row = lds + (colbase + lo) * 32 + hi * 16;
  v8h p0 = *(const v8h*)(row);
  v8h p1 = *(const v8h*)(row + 8);
  return frag_cat(p0, p1);
}

// Convert 16 consecutive f32 (64B-aligned) to 16 halves in LDS (two b128 stores).
static __device__ __forceinline__ void stage16(const float* __restrict__ src,
                                               _Float16* dst) {
  const float4* s4 = (const float4*)src;
  float4 f0 = s4[0], f1 = s4[1], f2 = s4[2], f3 = s4[3];
  v8h h0, h1;
  h0[0] = (_Float16)f0.x; h0[1] = (_Float16)f0.y; h0[2] = (_Float16)f0.z; h0[3] = (_Float16)f0.w;
  h0[4] = (_Float16)f1.x; h0[5] = (_Float16)f1.y; h0[6] = (_Float16)f1.z; h0[7] = (_Float16)f1.w;
  h1[0] = (_Float16)f2.x; h1[1] = (_Float16)f2.y; h1[2] = (_Float16)f2.z; h1[3] = (_Float16)f2.w;
  h1[4] = (_Float16)f3.x; h1[5] = (_Float16)f3.y; h1[6] = (_Float16)f3.z; h1[7] = (_Float16)f3.w;
  *(v8h*)(dst)     = h0;
  *(v8h*)(dst + 8) = h1;
}

#define WMMA_F16(a, b, c) \
  __builtin_amdgcn_wmma_f32_16x16x32_f16(false, (a), false, (b), (short)0, (c), false, false)

// ---------------------------------------------------------------------------
// Kernel 1: xz = x @ in_w^T ; split columns [0,1024) -> xc, [1024,2048) -> z.
// Block = 128 threads (4 waves), tile 64x64, K-step 32. dir=1 reads x flipped
// in time so the whole downstream pipeline runs in flipped coordinates.
// ---------------------------------------------------------------------------
__global__ __launch_bounds__(128)
void k_gemm_in(const float* __restrict__ x, const float* __restrict__ in_w,
               float* __restrict__ xc, float* __restrict__ z, int dir) {
  __shared__ _Float16 sA[64 * 32];
  __shared__ _Float16 sB[64 * 32];
  const int t = threadIdx.x;
  const int lane = t & 31, wave = t >> 5;
  const int wm = wave >> 1, wn = wave & 1;
  const int m0 = blockIdx.x * 64;
  const int n0 = blockIdx.y * 64;

  float* dst; int nc0;
  if (n0 >= DI) { dst = z; nc0 = n0 - DI; } else { dst = xc; nc0 = n0; }

  const int arow = t >> 1, aseg = t & 1;
  const int gm = m0 + arow;
  const int b = gm >> 11, s = gm & (SEQ - 1);
  const int srow = dir ? ((b << 11) + (SEQ - 1 - s)) : gm;
  const float* aptr = x + (size_t)srow * DM + aseg * 16;
  const float* bptr = in_w + (size_t)(n0 + arow) * DM + aseg * 16;

  v8f acc[2][2] = {};

  for (int k0 = 0; k0 < DM; k0 += 32) {
    stage16(aptr + k0, &sA[arow * 32 + aseg * 16]);
    stage16(bptr + k0, &sB[arow * 32 + aseg * 16]);
    if (k0 + 32 < DM) {
      __builtin_prefetch(aptr + k0 + 32, 0, 3);   // global_prefetch_b8, near
      __builtin_prefetch(bptr + k0 + 32, 0, 3);
    }
    __syncthreads();
    v16h af[2], bf[2];
    af[0] = frag_A(sA, wm * 32,      lane);
    af[1] = frag_A(sA, wm * 32 + 16, lane);
    bf[0] = frag_B(sB, wn * 32,      lane);
    bf[1] = frag_B(sB, wn * 32 + 16, lane);
#pragma unroll
    for (int am = 0; am < 2; ++am)
#pragma unroll
      for (int bn = 0; bn < 2; ++bn)
        acc[am][bn] = WMMA_F16(af[am], bf[bn], acc[am][bn]);
    __syncthreads();
  }

  const int lo = lane & 15, hi = lane >> 4;
#pragma unroll
  for (int am = 0; am < 2; ++am)
#pragma unroll
    for (int bn = 0; bn < 2; ++bn) {
      const int col = nc0 + wn * 32 + bn * 16 + lo;
#pragma unroll
      for (int r = 0; r < 8; ++r) {
        const int m = m0 + wm * 32 + am * 16 + r + hi * 8;
        dst[(size_t)m * DI + col] = acc[am][bn][r];
      }
    }
}

// ---------------------------------------------------------------------------
// Kernel 2: depthwise causal conv (width 4) + bias + SiLU.
// ---------------------------------------------------------------------------
__global__ __launch_bounds__(256)
void k_conv_silu(const float* __restrict__ xc, const float* __restrict__ cw,
                 const float* __restrict__ cb, float* __restrict__ U) {
  const int idx = blockIdx.x * 256 + threadIdx.x;      // < BS*DI
  const int d = idx & (DI - 1);
  const int r = idx >> 10;
  const int s = r & (SEQ - 1);
  float acc = cb[d];
#pragma unroll
  for (int j = 0; j < 4; ++j) {
    const int ss = s - 3 + j;
    if (ss >= 0) acc += cw[d * 4 + j] * xc[(size_t)(r - 3 + j) * DI + d];
  }
  U[idx] = acc / (1.0f + __expf(-acc));                // silu
}

// ---------------------------------------------------------------------------
// Kernel 3: xdb = u @ xproj_w^T  (M=4096, N=48, K=1024).
// Block = 128 threads (4 waves), tile M=64 x N=48; each wave owns 16 rows.
// ---------------------------------------------------------------------------
__global__ __launch_bounds__(128)
void k_gemm_xproj(const float* __restrict__ U, const float* __restrict__ W,
                  float* __restrict__ xdb) {
  __shared__ _Float16 sA[64 * 32];
  __shared__ _Float16 sB[48 * 32];
  const int t = threadIdx.x;
  const int lane = t & 31, wave = t >> 5;
  const int m0 = blockIdx.x * 64;
  const int arow = t >> 1, aseg = t & 1;
  const float* aptr = U + (size_t)(m0 + arow) * DI + aseg * 16;
  const float* bptr = W + (size_t)arow * DI + aseg * 16;   // valid for t<96

  v8f acc[3] = {};

  for (int k0 = 0; k0 < DI; k0 += 32) {
    stage16(aptr + k0, &sA[arow * 32 + aseg * 16]);
    if (t < 96) stage16(bptr + k0, &sB[arow * 32 + aseg * 16]);
    if (k0 + 32 < DI) __builtin_prefetch(aptr + k0 + 32, 0, 3);
    __syncthreads();
    v16h a = frag_A(sA, wave * 16, lane);
#pragma unroll
    for (int nt = 0; nt < 3; ++nt) {
      v16h bf = frag_B(sB, nt * 16, lane);
      acc[nt] = WMMA_F16(a, bf, acc[nt]);
    }
    __syncthreads();
  }

  const int lo = lane & 15, hi = lane >> 4;
#pragma unroll
  for (int nt = 0; nt < 3; ++nt)
#pragma unroll
    for (int r = 0; r < 8; ++r) {
      const int m = m0 + wave * 16 + r + hi * 8;
      xdb[(size_t)m * 48 + nt * 16 + lo] = acc[nt][r];
    }
}

// ---------------------------------------------------------------------------
// TDM chunk loader: DMA CHUNK rows x 48 floats of xdb into LDS.
// D# per CDNA5 ISA: 2-D tensor, data_size=4B,
// tensor_dim0 = 48, tile 48 x CHUNK, dim0 stride 48. Issued by wave 0 only
// (uniform branch; tensor ops ignore EXEC but waves 1..N-1 branch around it).
// ---------------------------------------------------------------------------
static __device__ __forceinline__ void scan_load_chunk(const float* __restrict__ src,
                                                       float* __restrict__ dstLds,
                                                       int tid) {
#if MAMBA_TDM
  if ((tid >> 5) == 0) {
    const unsigned long long ga = (unsigned long long)(size_t)src;
    u32x4 g0;
    g0[0] = 1u;                                            // count=1 (valid user D#)
    g0[1] = (unsigned)(size_t)dstLds;                      // lds_addr (low 32 of flat)
    g0[2] = (unsigned)(ga & 0xFFFFFFFFull);                // global_addr[31:0]
    g0[3] = (unsigned)((ga >> 32) & 0x01FFFFFFull)         // global_addr[56:32]
          | 0x80000000u;                                   // type=2 ("image")
    i32x8 g1;
    g1[0] = (int)(2u << 16);          // workgroup_mask=0, data_size=4B
    g1[1] = (int)(48u << 16);         // atomic_barrier=0, tensor_dim0[15:0]=48
    g1[2] = (int)((unsigned)CHUNK << 16); // tensor_dim0 hi=0, tensor_dim1[15:0]=CHUNK
    g1[3] = (int)(48u << 16);         // tensor_dim1 hi=0, tile_dim0=48
    g1[4] = CHUNK;                    // tile_dim1=CHUNK, tile_dim2=0
    g1[5] = 48;                       // tensor_dim0_stride = 48 elements
    g1[6] = 0;
    g1[7] = 0;
    i32x4 gz4 = (i32x4)0;
#if defined(__clang_major__) && (__clang_major__ >= 23)
    i32x8 gz8 = (i32x8)0;
    __builtin_amdgcn_tensor_load_to_lds(g0, g1, gz4, gz4, gz8, 0);
#else
    __builtin_amdgcn_tensor_load_to_lds(g0, g1, gz4, gz4, 0);
#endif
  }
#else
  for (int i = tid; i < CHUNK * 48; i += 1024) dstLds[i] = src[i];
#endif
}

static __device__ __forceinline__ void scan_wait_chunk() {
#if MAMBA_TDM
  __builtin_amdgcn_s_wait_tensorcnt(0);
#endif
}

// ---------------------------------------------------------------------------
// Kernel 4: selective scan. 1 block per batch, 1 thread per channel (1024).
// h[16] kept in registers; xdb rows streamed into LDS by the TDM in
// double-buffered CHUNK-row tiles: one DMA + one s_wait_tensorcnt + one
// barrier per CHUNK timesteps (vs per-step broadcast).
// dt = softplus(xdb[:16] . dt_w[d] + dt_b[d]); g = (y + u*D) * silu(z).
// ---------------------------------------------------------------------------
__global__ __launch_bounds__(1024)
void k_scan(const float* __restrict__ xdb, const float* __restrict__ U,
            const float* __restrict__ Z, const float* __restrict__ dt_w,
            const float* __restrict__ dt_b, const float* __restrict__ A_log,
            const float* __restrict__ Dp, float* __restrict__ G) {
  __shared__ __align__(16) float sbuf[2][CHUNK * 48];
  const int tid = threadIdx.x;
  const int b = blockIdx.x;
  const int d = tid;

  float dtw[16], Af[16], h[16];
#pragma unroll
  for (int i = 0; i < 16; ++i) {
    dtw[i] = dt_w[d * 16 + i];
    Af[i]  = -__expf(A_log[d * 16 + i]);
    h[i]   = 0.0f;
  }
  const float dtb = dt_b[d];
  const float Dv  = Dp[d];

  const float* xrow = xdb + (size_t)(b << 11) * 48;

  scan_load_chunk(xrow, sbuf[0], tid);       // prologue: chunk 0
  scan_wait_chunk();
  __syncthreads();

  for (int c = 0; c < SEQ / CHUNK; ++c) {
    if (c + 1 < SEQ / CHUNK)                 // prefetch next chunk via TDM
      scan_load_chunk(xrow + (size_t)(c + 1) * CHUNK * 48, sbuf[(c + 1) & 1], tid);

    const float* sh = sbuf[c & 1];
    for (int t2 = 0; t2 < CHUNK; ++t2) {
      const int r = (b << 11) + c * CHUNK + t2;
      const float* row = sh + t2 * 48;

      float dtr = dtb;
#pragma unroll
      for (int i = 0; i < 16; ++i) dtr += row[i] * dtw[i];
      const float dt = (dtr > 20.0f) ? dtr : log1pf(__expf(dtr));

      const float u  = U[(size_t)r * DI + d];
      const float zv = Z[(size_t)r * DI + d];
      const float du = dt * u;

      float y = 0.0f;
#pragma unroll
      for (int n = 0; n < 16; ++n) {
        h[n] = __expf(dt * Af[n]) * h[n] + du * row[16 + n];
        y += h[n] * row[32 + n];
      }
      const float sz = zv / (1.0f + __expf(-zv));
      G[(size_t)r * DI + d] = (y + u * Dv) * sz;
    }
    scan_wait_chunk();                       // next chunk landed
    __syncthreads();
  }
}

// ---------------------------------------------------------------------------
// Kernel 5: ym = g @ out_w^T  (M=4096, N=256, K=1024). Same tiling as k_gemm_in.
// ---------------------------------------------------------------------------
__global__ __launch_bounds__(128)
void k_gemm_out(const float* __restrict__ G, const float* __restrict__ W,
                float* __restrict__ ym) {
  __shared__ _Float16 sA[64 * 32];
  __shared__ _Float16 sB[64 * 32];
  const int t = threadIdx.x;
  const int lane = t & 31, wave = t >> 5;
  const int wm = wave >> 1, wn = wave & 1;
  const int m0 = blockIdx.x * 64;
  const int n0 = blockIdx.y * 64;
  const int arow = t >> 1, aseg = t & 1;
  const float* aptr = G + (size_t)(m0 + arow) * DI + aseg * 16;
  const float* bptr = W + (size_t)(n0 + arow) * DI + aseg * 16;

  v8f acc[2][2] = {};

  for (int k0 = 0; k0 < DI; k0 += 32) {
    stage16(aptr + k0, &sA[arow * 32 + aseg * 16]);
    stage16(bptr + k0, &sB[arow * 32 + aseg * 16]);
    if (k0 + 32 < DI) {
      __builtin_prefetch(aptr + k0 + 32, 0, 3);
      __builtin_prefetch(bptr + k0 + 32, 0, 3);
    }
    __syncthreads();
    v16h af[2], bf[2];
    af[0] = frag_A(sA, wm * 32,      lane);
    af[1] = frag_A(sA, wm * 32 + 16, lane);
    bf[0] = frag_B(sB, wn * 32,      lane);
    bf[1] = frag_B(sB, wn * 32 + 16, lane);
#pragma unroll
    for (int am = 0; am < 2; ++am)
#pragma unroll
      for (int bn = 0; bn < 2; ++bn)
        acc[am][bn] = WMMA_F16(af[am], bf[bn], acc[am][bn]);
    __syncthreads();
  }

  const int lo = lane & 15, hi = lane >> 4;
#pragma unroll
  for (int am = 0; am < 2; ++am)
#pragma unroll
    for (int bn = 0; bn < 2; ++bn) {
      const int col = n0 + wn * 32 + bn * 16 + lo;
#pragma unroll
      for (int r = 0; r < 8; ++r) {
        const int m = m0 + wm * 32 + am * 16 + r + hi * 8;
        ym[(size_t)m * DM + col] = acc[am][bn][r];
      }
    }
}

// ---------------------------------------------------------------------------
// Kernel 6: residual + RMSNorm + (un)flip + write into the concat output.
// One 256-thread block per output row; wave32 shfl_xor butterfly reduction.
// ---------------------------------------------------------------------------
__global__ __launch_bounds__(256)
void k_resnorm(const float* __restrict__ ym, const float* __restrict__ x,
               const float* __restrict__ w, float* __restrict__ out, int dir) {
  __shared__ float part[8];
  const int t = threadIdx.x, lane = t & 31, wid = t >> 5;
  const int row = blockIdx.x;                 // output position b*SEQ + s
  const int b = row >> 11, s = row & (SEQ - 1);
  const int srow = dir ? ((b << 11) + (SEQ - 1 - s)) : row;  // flipped source

  const float v = ym[(size_t)srow * DM + t] + x[(size_t)row * DM + t];

  float ss = v * v;
#pragma unroll
  for (int m = 16; m > 0; m >>= 1) ss += __shfl_xor(ss, m, 32);
  if (lane == 0) part[wid] = ss;
  __syncthreads();
  float tot = 0.0f;
#pragma unroll
  for (int i = 0; i < 8; ++i) tot += part[i];

  const float rms = sqrtf(tot * (1.0f / DM) + 1e-5f);
  out[(size_t)row * (2 * DM) + dir * DM + t] = v / rms * w[t];
}

// ---------------------------------------------------------------------------
extern "C" void kernel_launch(void* const* d_in, const int* in_sizes, int n_in,
                              void* d_out, int out_size, void* d_ws, size_t ws_size,
                              hipStream_t stream) {
  (void)in_sizes; (void)n_in; (void)out_size; (void)ws_size;
  const float* x = (const float*)d_in[0];
  float* out = (float*)d_out;

  float* ws  = (float*)d_ws;
  float* XC  = ws;                               // BS*DI  (reused as G after scan)
  float* Z   = XC  + (size_t)BS * DI;            // BS*DI
  float* U   = Z   + (size_t)BS * DI;            // BS*DI
  float* XDB = U   + (size_t)BS * DI;            // BS*48
  float* YM  = XDB + (size_t)BS * 48;            // BS*DM

  for (int dir = 0; dir < 2; ++dir) {
    const int base = 1 + dir * 9;
    const float* in_w = (const float*)d_in[base + 0];
    const float* cw   = (const float*)d_in[base + 1];
    const float* cb   = (const float*)d_in[base + 2];
    const float* xpw  = (const float*)d_in[base + 3];
    const float* dtw  = (const float*)d_in[base + 4];
    const float* dtb  = (const float*)d_in[base + 5];
    const float* Alog = (const float*)d_in[base + 6];
    const float* Dp   = (const float*)d_in[base + 7];
    const float* outw = (const float*)d_in[base + 8];
    const float* nw   = (const float*)d_in[19 + dir];

    k_gemm_in   <<<dim3(BS / 64, (2 * DI) / 64), dim3(128), 0, stream>>>(x, in_w, XC, Z, dir);
    k_conv_silu <<<dim3(BS * DI / 256),          dim3(256), 0, stream>>>(XC, cw, cb, U);
    k_gemm_xproj<<<dim3(BS / 64),                dim3(128), 0, stream>>>(U, xpw, XDB);
    k_scan      <<<dim3(BATCH),                  dim3(1024), 0, stream>>>(XDB, U, Z, dtw, dtb, Alog, Dp, XC);
    k_gemm_out  <<<dim3(BS / 64, DM / 64),       dim3(128), 0, stream>>>(XC, outw, YM);
    k_resnorm   <<<dim3(BS),                     dim3(256), 0, stream>>>(YM, x, nw, out, dir);
  }
}
